// VLMMultiHeadAttention_21277267984719
// MI455X (gfx1250) — matrix operand order repaired and checked by reading, not compiled, not run on hardware
//
#include <hip/hip_runtime.h>

typedef __bf16 bf16_t;
typedef bf16_t v8bf  __attribute__((ext_vector_type(8)));
typedef bf16_t v16bf __attribute__((ext_vector_type(16)));
typedef float  v8f   __attribute__((ext_vector_type(8)));
typedef int    v4i   __attribute__((ext_vector_type(4)));
typedef unsigned short u16;
typedef u16    v8us  __attribute__((ext_vector_type(8)));

// Problem constants (match reference)
constexpr int Bv = 8, Tt = 1221, Cc = 1024, Hh = 16, Dd = 64;
constexpr int Mrows = Bv * Tt;            // 9768
constexpr int Mpad  = 9856;               // 77 * 128, lets async tile loads run unguarded
constexpr int TP = 1232;                  // padded T stride for V^T (16B-aligned rows)
constexpr int HV = 6, HVT = 12;           // int(16*0.4)=6 vision heads, 6 text heads
constexpr float LN_THETA = 9.210340371976184f;  // ln(10000)

#if defined(__has_builtin)
#if __has_builtin(__builtin_amdgcn_global_load_async_to_lds_b128)
#define HAVE_ASYNC_LDS 1
#endif
#endif
#ifndef HAVE_ASYNC_LDS
#define HAVE_ASYNC_LDS 0
#endif

#if HAVE_ASYNC_LDS
typedef __attribute__((address_space(1))) v4i* gas_v4i;   // global src
typedef __attribute__((address_space(3))) v4i* lds_v4i;   // LDS dst
__device__ __forceinline__ void async_copy16(const u16* gsrc, u16* ldst) {
  __builtin_amdgcn_global_load_async_to_lds_b128(
      (gas_v4i)(u16*)gsrc, (lds_v4i)ldst, 0, 0);
}
__device__ __forceinline__ void wait_asynccnt0() {
#if __has_builtin(__builtin_amdgcn_s_wait_asynccnt)
  __builtin_amdgcn_s_wait_asynccnt(0);
#else
  asm volatile("s_wait_asynccnt 0" ::: "memory");
#endif
}
#else
__device__ __forceinline__ void async_copy16(const u16* gsrc, u16* ldst) {
  *(v8us*)ldst = *(const v8us*)gsrc;     // sync fallback
}
__device__ __forceinline__ void wait_asynccnt0() {}
#endif

// native bf16 convert (emits v_cvt_pk_bf16_f32)
__device__ __forceinline__ u16 f2bf(float x) {
  __bf16 h = (__bf16)x;
  return __builtin_bit_cast(u16, h);
}

__device__ __forceinline__ v16bf cat16(v8bf lo, v8bf hi) {
  return __builtin_shufflevector(lo, hi, 0,1,2,3,4,5,6,7,8,9,10,11,12,13,14,15);
}
// A/B fragment in this code's k = 16h + s order: 16 contiguous bf16 per lane
__device__ __forceinline__ v16bf ld16_contig(const u16* p) {
  return cat16(*(const v8bf*)p, *(const v8bf*)(p + 8));
}
// A-style fragment: two 8-elem chunks separated by 16 (K and K+16)
__device__ __forceinline__ v16bf ld16_split(const u16* p) {
  return cat16(*(const v8bf*)p, *(const v8bf*)(p + 16));
}
__device__ __forceinline__ v8f vzero8() {
  v8f z;
#pragma unroll
  for (int i = 0; i < 8; ++i) z[i] = 0.f;
  return z;
}
__device__ __forceinline__ v8f wmma_bf16(v16bf a, v16bf b, v8f c) {
  return __builtin_amdgcn_wmma_f32_16x16x32_bf16(false, a, false, b, (short)0, c, false, false);
}

// ---------------------------------------------------------------------------
// Kernel 0: f32 -> bf16 conversion (x and weights), zero-fill [n, npad)
// ---------------------------------------------------------------------------
__global__ __launch_bounds__(256)
void cvt_bf16_kernel(const float* __restrict__ src, u16* __restrict__ dst,
                     int n, int npad) {
  const int i = (blockIdx.x * 256 + threadIdx.x) * 4;
  if (i >= npad) return;
  if (i + 4 <= n) {
    float4 f = *(const float4*)(src + i);
    dst[i + 0] = f2bf(f.x);
    dst[i + 1] = f2bf(f.y);
    dst[i + 2] = f2bf(f.z);
    dst[i + 3] = f2bf(f.w);
  } else {
#pragma unroll
    for (int j = 0; j < 4; ++j) {
      const int e = i + j;
      if (e < npad) dst[e] = (e < n) ? f2bf(src[e]) : (u16)0;
    }
  }
}

// ---------------------------------------------------------------------------
// Kernel 1: Q/K/V projection (xb[Mpad,C]bf16 @ Wb[N,C]^T) + fused RoPE on Q,K.
// Double-buffered async-DMA LDS tiles. blockIdx.z selects {Q,K,V}.
// Q,K -> bf16 [B,H,T,D]; V -> bf16 [B,H,D,TP] (transposed).
// ---------------------------------------------------------------------------
__global__ __launch_bounds__(256)
void qkv_rope_kernel(const u16* __restrict__ xb,
                     const u16* __restrict__ Wqb,
                     const u16* __restrict__ Wkb,
                     const u16* __restrict__ Wvb,
                     u16* __restrict__ Qb, u16* __restrict__ Kb, u16* __restrict__ Vt) {
  __shared__ u16 lA[2][128][40];   // x tile  [m][k], double buffered
  __shared__ u16 lB[2][128][40];   // W tile  [n][k]

  const int z = blockIdx.z;
  const u16* W = (z == 0) ? Wqb : ((z == 1) ? Wkb : Wvb);
  u16* dst = (z == 0) ? Qb : ((z == 1) ? Kb : Vt);
  const bool rope = (z < 2);

  const int tid = threadIdx.x;
  const int m0 = blockIdx.x * 128;
  const int n0 = blockIdx.y * 128;
  const int wave = tid >> 5, ln = tid & 31;
  const int wm = wave >> 1, wn = wave & 1;          // 4x2 wave grid
  const int lane16 = ln & 15;
  const int halfAdd = (ln >> 4) << 3;               // +8 rows for upper half
  const int lrow = tid >> 1;
  const int lcol = (tid & 1) * 16;

  v8f acc[2][4];
#pragma unroll
  for (int i = 0; i < 2; ++i)
#pragma unroll
    for (int j = 0; j < 4; ++j) acc[i][j] = vzero8();

  auto issue = [&](int k0, int buf) {
    const u16* gA = xb + (long)(m0 + lrow) * Cc + k0 + lcol;   // Mpad-padded: always in bounds
    const u16* gB = W  + (long)(n0 + lrow) * Cc + k0 + lcol;
    async_copy16(gA,     &lA[buf][lrow][lcol]);
    async_copy16(gA + 8, &lA[buf][lrow][lcol + 8]);
    async_copy16(gB,     &lB[buf][lrow][lcol]);
    async_copy16(gB + 8, &lB[buf][lrow][lcol + 8]);
  };

  issue(0, 0);
  wait_asynccnt0();
  __syncthreads();

  int cur = 0;
  for (int k0 = 0; k0 < Cc; k0 += 32) {
    if (k0 + 32 < Cc) issue(k0 + 32, cur ^ 1);   // overlap DMA with WMMA

    v16bf afr[2], bfr[4];
#pragma unroll
    for (int mi = 0; mi < 2; ++mi) {
      const int row = wm * 32 + mi * 16 + lane16;
      const int kc = (ln < 16) ? 0 : 16;
      afr[mi] = ld16_contig(&lA[cur][row][kc]);
    }
#pragma unroll
    for (int ni = 0; ni < 4; ++ni) {
      const int row = wn * 64 + ni * 16 + lane16;
      const int kc = (ln < 16) ? 0 : 16;
      bfr[ni] = ld16_contig(&lB[cur][row][kc]);
    }
#pragma unroll
    for (int mi = 0; mi < 2; ++mi)
#pragma unroll
      for (int ni = 0; ni < 4; ++ni)
        acc[mi][ni] = wmma_bf16(afr[mi], bfr[ni], acc[mi][ni]);

    wait_asynccnt0();
    __syncthreads();
    cur ^= 1;
  }

  // --- epilogue: RoPE (pairs are adjacent lanes) + bf16 stores ---
#pragma unroll
  for (int mi = 0; mi < 2; ++mi) {
#pragma unroll
    for (int ni = 0; ni < 4; ++ni) {
#pragma unroll
      for (int r = 0; r < 8; ++r) {
        float v = acc[mi][ni][r];
        const int gm = m0 + wm * 32 + mi * 16 + r + halfAdd;
        const int gn = n0 + wn * 64 + ni * 16 + lane16;
        float other = __shfl_xor(v, 1, 32);  // uniform: all lanes execute
        float outv = v;
        if (rope) {
          const int dh = gn & 63;
          const int tpos = gm % Tt;
          // inv_freq = theta^(-(dh & ~1)/64) = exp(-ln(theta) * (dh>>1)/32)
          const float inv = __expf(-LN_THETA * (float)(dh >> 1) * (1.0f / 32.0f));
          float sn, cs;
          __sincosf((float)tpos * inv, &sn, &cs);
          outv = (dh & 1) ? (other * sn + v * cs)   // imag: xr*sin + xi*cos
                          : (v * cs - other * sn);  // real: xr*cos - xi*sin
        }
        if (gm < Mrows) {
          const int b = gm / Tt, t = gm % Tt, h = gn >> 6, dh = gn & 63;
          const long idx = (z == 2)
              ? (((long)(b * Hh + h)) * Dd + dh) * TP + t     // V transposed
              : (((long)(b * Hh + h)) * Tt + t) * Dd + dh;    // Q/K row-major
          dst[idx] = f2bf(outv);
        }
      }
    }
  }
}

// ---------------------------------------------------------------------------
// Modality mask (head_V | head_T | head_VT)
// ---------------------------------------------------------------------------
__device__ __forceinline__ bool valid_qk(int h, int q, int k, int S_V) {
  if (k >= Tt || q >= Tt) return false;
  if (h < HV)  return (q < S_V) && (k < S_V);
  if (h < HVT) return (q >= S_V) && (k >= S_V) && (q >= k);
  return (k < S_V) || (q >= k);
}

// ---------------------------------------------------------------------------
// Kernel 2: flash attention. Block = (128 q rows) x (b,h). Each wave owns 16
// q rows; K/V streamed in double-buffered 32-key async LDS tiles.
// ---------------------------------------------------------------------------
__global__ __launch_bounds__(256)
void attn_kernel(const u16* __restrict__ Qb, const u16* __restrict__ Kb,
                 const u16* __restrict__ Vt, const int* __restrict__ svp,
                 u16* __restrict__ Ob) {
  __shared__ u16 lK[2][32][72];     // [key][d]   -> QK^T B-fragment contiguous
  __shared__ u16 lV[2][64][40];     // [d][key]   -> PV  B-fragment contiguous
  __shared__ u16 lP[8][16][40];     // per-wave P tile [q][k]

  const int S_V = *svp;
  const int tid = threadIdx.x, wave = tid >> 5, ln = tid & 31;
  const int h = blockIdx.y, b = blockIdx.z;
  const int q0 = blockIdx.x * 128 + wave * 16;
  const long baseBH = ((long)(b * Hh + h)) * Tt * Dd;   // Q/K base
  const long baseV  = ((long)(b * Hh + h)) * Dd * TP;   // V^T base
  const int lane16 = ln & 15;
  const int halfAdd = (ln >> 4) << 3;

  // Q fragments (2 x 16x32 along D=64), rows clamped for tail tiles
  v16bf qf[2];
  {
    int t = q0 + lane16; if (t >= Tt) t = Tt - 1;
    const u16* qp = Qb + baseBH + (long)t * Dd;
#pragma unroll
    for (int dc = 0; dc < 2; ++dc) {
      const int d0q = dc * 32 + ((ln < 16) ? 0 : 16);
      qf[dc] = cat16(*(const v8bf*)(qp + d0q), *(const v8bf*)(qp + d0q + 8));
    }
  }

  float mrow[8], lsum[8];
  v8f oacc[4];
#pragma unroll
  for (int r = 0; r < 8; ++r) { mrow[r] = -1e30f; lsum[r] = 0.f; }
#pragma unroll
  for (int dt = 0; dt < 4; ++dt) oacc[dt] = vzero8();

  const int key = tid >> 3, d0 = (tid & 7) * 8;   // K tile coords (32x64)
  const int dv = tid >> 2, k0v = (tid & 3) * 8;   // V tile coords (64x32)

  auto issue_tile = [&](int kb, int buf) {
    if (kb + 32 <= Tt) {
      async_copy16(Kb + baseBH + (long)(kb + key) * Dd + d0, &lK[buf][key][d0]);
      async_copy16(Vt + baseV + (long)dv * TP + kb + k0v,    &lV[buf][dv][k0v]);
    } else {
      // ragged tail tile: per-element guarded loads
#pragma unroll
      for (int j = 0; j < 8; ++j) {
        const int t = kb + key;
        lK[buf][key][d0 + j] = (t < Tt) ? Kb[baseBH + (long)t * Dd + d0 + j] : (u16)0;
        const int tv = kb + k0v + j;
        lV[buf][dv][k0v + j] = (tv < Tt) ? Vt[baseV + (long)dv * TP + tv] : (u16)0;
      }
    }
  };

  issue_tile(0, 0);
  wait_asynccnt0();
  __syncthreads();

  int cur = 0;
  for (int kb = 0; kb < Tt; kb += 32) {
    if (kb + 32 < Tt) issue_tile(kb + 32, cur ^ 1);   // overlap DMA with compute

    // S = Q * K^T  (16 q x 32 keys)
    v8f sacc[2] = { vzero8(), vzero8() };
#pragma unroll
    for (int kc = 0; kc < 2; ++kc) {
#pragma unroll
      for (int dc = 0; dc < 2; ++dc) {
        const int krow = kc * 16 + lane16;
        const int dd = dc * 32 + ((ln < 16) ? 0 : 16);
        sacc[kc] = wmma_bf16(qf[dc], ld16_contig(&lK[cur][krow][dd]), sacc[kc]);
      }
    }

    // mask + online softmax (rows live in 16-lane halves)
    float alpha[8];
#pragma unroll
    for (int r = 0; r < 8; ++r) {
      const int q = q0 + r + halfAdd;
      const int k0i = kb + lane16, k1i = kb + 16 + lane16;
      float s0 = valid_qk(h, q, k0i, S_V) ? sacc[0][r] * 0.125f : -1e30f;
      float s1 = valid_qk(h, q, k1i, S_V) ? sacc[1][r] * 0.125f : -1e30f;
      float mx = fmaxf(s0, s1);
#pragma unroll
      for (int off = 8; off >= 1; off >>= 1) mx = fmaxf(mx, __shfl_xor(mx, off, 32));
      const float mnew = fmaxf(mrow[r], mx);
      const float a = __expf(mrow[r] - mnew);
      const float p0 = __expf(s0 - mnew), p1 = __expf(s1 - mnew);
      float rs = p0 + p1;
#pragma unroll
      for (int off = 8; off >= 1; off >>= 1) rs += __shfl_xor(rs, off, 32);
      lsum[r] = lsum[r] * a + rs;
      mrow[r] = mnew;
      alpha[r] = a;
      lP[wave][r + halfAdd][lane16]      = f2bf(p0);
      lP[wave][r + halfAdd][16 + lane16] = f2bf(p1);
    }
#pragma unroll
    for (int dt = 0; dt < 4; ++dt)
#pragma unroll
      for (int r = 0; r < 8; ++r) oacc[dt][r] = oacc[dt][r] * alpha[r];

    // wave-local LDS RAW on lP: split dependency counter wait (CDNA5)
    asm volatile("s_wait_dscnt 0" ::: "memory");

    // O += P * V
    {
      const int kc0 = (ln < 16) ? 0 : 16;
      v16bf pf = ld16_contig(&lP[wave][lane16][kc0]);
#pragma unroll
      for (int dt = 0; dt < 4; ++dt) {
        const int drow = dt * 16 + lane16;
        const int kk = (ln < 16) ? 0 : 16;
        oacc[dt] = wmma_bf16(pf, ld16_contig(&lV[cur][drow][kk]), oacc[dt]);
      }
    }

    wait_asynccnt0();
    __syncthreads();
    cur ^= 1;
  }

  // normalize + store attention output as bf16 [Mpad, C] with c = h*64+d
#pragma unroll
  for (int dt = 0; dt < 4; ++dt) {
#pragma unroll
    for (int r = 0; r < 8; ++r) {
      const int q = q0 + r + halfAdd;
      if (q < Tt) {
        const float denom = lsum[r];
        const float val = (mrow[r] > -1e29f && denom > 0.f) ? oacc[dt][r] / denom : 0.f;
        const int d = dt * 16 + lane16;
        Ob[((long)(b * Tt + q)) * Cc + h * Dd + d] = f2bf(val);
      }
    }
  }
}

// ---------------------------------------------------------------------------
// Kernel 3: out = Ob[Mpad,C](bf16) @ Wob[N,C]^T + bo, f32 store.
// Same double-buffered async pipeline.
// ---------------------------------------------------------------------------
__global__ __launch_bounds__(256)
void proj_kernel(const u16* __restrict__ Ob, const u16* __restrict__ Wob,
                 const float* __restrict__ bo, float* __restrict__ out) {
  __shared__ u16 lA[2][128][40];
  __shared__ u16 lB[2][128][40];

  const int tid = threadIdx.x;
  const int m0 = blockIdx.x * 128;
  const int n0 = blockIdx.y * 128;
  const int wave = tid >> 5, ln = tid & 31;
  const int wm = wave >> 1, wn = wave & 1;
  const int lane16 = ln & 15;
  const int halfAdd = (ln >> 4) << 3;
  const int lrow = tid >> 1;
  const int lcol = (tid & 1) * 16;

  v8f acc[2][4];
#pragma unroll
  for (int i = 0; i < 2; ++i)
#pragma unroll
    for (int j = 0; j < 4; ++j) acc[i][j] = vzero8();

  auto issue = [&](int k0, int buf) {
    const u16* gA = Ob  + (long)(m0 + lrow) * Cc + k0 + lcol;   // Mpad rows: in bounds
    const u16* gB = Wob + (long)(n0 + lrow) * Cc + k0 + lcol;
    async_copy16(gA,     &lA[buf][lrow][lcol]);
    async_copy16(gA + 8, &lA[buf][lrow][lcol + 8]);
    async_copy16(gB,     &lB[buf][lrow][lcol]);
    async_copy16(gB + 8, &lB[buf][lrow][lcol + 8]);
  };

  issue(0, 0);
  wait_asynccnt0();
  __syncthreads();

  int cur = 0;
  for (int k0 = 0; k0 < Cc; k0 += 32) {
    if (k0 + 32 < Cc) issue(k0 + 32, cur ^ 1);

    v16bf afr[2], bfr[4];
#pragma unroll
    for (int mi = 0; mi < 2; ++mi) {
      const int row = wm * 32 + mi * 16 + lane16;
      const int kc = (ln < 16) ? 0 : 16;
      afr[mi] = ld16_contig(&lA[cur][row][kc]);
    }
#pragma unroll
    for (int ni = 0; ni < 4; ++ni) {
      const int row = wn * 64 + ni * 16 + lane16;
      const int kc = (ln < 16) ? 0 : 16;
      bfr[ni] = ld16_contig(&lB[cur][row][kc]);
    }
#pragma unroll
    for (int mi = 0; mi < 2; ++mi)
#pragma unroll
      for (int ni = 0; ni < 4; ++ni)
        acc[mi][ni] = wmma_bf16(afr[mi], bfr[ni], acc[mi][ni]);

    wait_asynccnt0();
    __syncthreads();
    cur ^= 1;
  }

#pragma unroll
  for (int mi = 0; mi < 2; ++mi) {
#pragma unroll
    for (int ni = 0; ni < 4; ++ni) {
#pragma unroll
      for (int r = 0; r < 8; ++r) {
        const int gm = m0 + wm * 32 + mi * 16 + r + halfAdd;
        const int gn = n0 + wn * 64 + ni * 16 + lane16;
        if (gm < Mrows) out[(long)gm * Cc + gn] = acc[mi][ni][r] + bo[gn];
      }
    }
  }
}

// ---------------------------------------------------------------------------
extern "C" void kernel_launch(void* const* d_in, const int* in_sizes, int n_in,
                              void* d_out, int out_size, void* d_ws, size_t ws_size,
                              hipStream_t stream) {
  (void)in_sizes; (void)n_in; (void)out_size; (void)ws_size;
  const float* x  = (const float*)d_in[0];
  const float* Wq = (const float*)d_in[1];
  const float* Wk = (const float*)d_in[2];
  const float* Wv = (const float*)d_in[3];
  const float* Wo = (const float*)d_in[4];
  const float* bo = (const float*)d_in[5];
  const int*   sv = (const int*)d_in[6];

  char* ws = (char*)d_ws;
  const size_t segMp = (size_t)Mpad * Cc * sizeof(u16);           // 20,185,088 B
  const size_t segW  = (size_t)Cc * Cc * sizeof(u16);             //  2,097,152 B
  const size_t segQK = (size_t)Mrows * Cc * sizeof(u16);          // 20,004,864 B
  const size_t segV  = (size_t)Bv * Hh * Dd * TP * sizeof(u16);   // 20,185,088 B

  u16* xb  = (u16*)(ws);
  u16* Wqb = (u16*)(ws + segMp);
  u16* Wkb = (u16*)(ws + segMp + segW);
  u16* Wvb = (u16*)(ws + segMp + 2 * segW);
  u16* Wob = (u16*)(ws + segMp + 3 * segW);
  u16* Qb  = (u16*)(ws + segMp + 4 * segW);
  u16* Kb  = (u16*)(ws + segMp + 4 * segW + segQK);
  u16* Vt  = (u16*)(ws + segMp + 4 * segW + 2 * segQK);
  u16* Ob  = (u16*)(ws + segMp + 4 * segW + 2 * segQK + segV);

  // stage 0: bf16 conversions
  {
    const int nx = Mrows * Cc, nxp = Mpad * Cc;
    cvt_bf16_kernel<<<dim3((nxp / 4 + 255) / 256), 256, 0, stream>>>(x, xb, nx, nxp);
    const int nw = Cc * Cc;
    dim3 gw((nw / 4 + 255) / 256);
    cvt_bf16_kernel<<<gw, 256, 0, stream>>>(Wq, Wqb, nw, nw);
    cvt_bf16_kernel<<<gw, 256, 0, stream>>>(Wk, Wkb, nw, nw);
    cvt_bf16_kernel<<<gw, 256, 0, stream>>>(Wv, Wvb, nw, nw);
    cvt_bf16_kernel<<<gw, 256, 0, stream>>>(Wo, Wob, nw, nw);
  }

  dim3 g1((Mpad + 127) / 128, Cc / 128, 3);
  qkv_rope_kernel<<<g1, 256, 0, stream>>>(xb, Wqb, Wkb, Wvb, Qb, Kb, Vt);

  dim3 g2((Tt + 127) / 128, Hh, Bv);
  attn_kernel<<<g2, 256, 0, stream>>>(Qb, Kb, Vt, sv, Ob);

  dim3 g3((Mpad + 127) / 128, Cc / 128);
  proj_kernel<<<g3, 256, 0, stream>>>(Ob, Wob, bo, (float*)d_out);
}
